// DSSLayer_33474975105486
// MI455X (gfx1250) — matrix operand order repaired and checked by reading, not compiled, not run on hardware
//
#include <hip/hip_runtime.h>
#include <hip/hip_bf16.h>

#define BB 8
#define HH 512
#define LL 4096
#define NN 64
#define CC 64   // chunk length
#define NC 64   // number of chunks (LL/CC)

typedef __attribute__((ext_vector_type(16))) _Float16 v16h;
typedef __attribute__((ext_vector_type(8)))  _Float16 v8h;
typedef __attribute__((ext_vector_type(8)))  float    v8f;
typedef __attribute__((ext_vector_type(4)))  float    v4f;

// ---------------------------------------------------------------------------
// WMMA fragment loaders for V_WMMA_F32_16X16X32_F16 (wave32), following the
// CDNA5 ISA VGPR layouts (CDNA5 ISA §7.12.2).  All loads are 16-byte aligned
// contiguous v8h vectors -> lower to ds_load_b128.
// A: 16x32 f16. lane L<16 -> row L, halves 0..7 = K 0..7, halves 8..15 = K 16..23
//               lane L>=16 -> row L-16, halves 0..7 = K 8..15, halves 8..15 = K 24..31
// B: 32x16 f16. lanes 0..15 -> K 0..7,16..23 (halves), lanes 16..31 -> K 8..15,24..31; col = lane&15
// C/D: 8 f32.   element r -> M = r + (lane<16 ? 0 : 8), N = lane&15
// ---------------------------------------------------------------------------
__device__ __forceinline__ v16h load_a_frag(const _Float16* m, int rowBase, int kk, int lane) {
    const int row = rowBase + (lane & 15);
    const int b0  = (lane < 16) ? 0 : 8;
    const _Float16* p = m + row * 64 + kk + b0;          // 16B aligned
    const v8h lo = *(const v8h*)(p);                     // K = kk+b0 .. +7
    const v8h hi = *(const v8h*)(p + 16);                // K = kk+16+b0 .. +7
    return __builtin_shufflevector(lo, hi, 0,1,2,3,4,5,6,7,8,9,10,11,12,13,14,15);
}

// B(K,N) stored N-major: value = m[(colBase+N)*64 + K]
//   T0^T: B[k][tau] = T0[tau][k];  Q: B[rho][n] = Q[n][rho]
__device__ __forceinline__ v16h load_b_frag(const _Float16* m, int colBase, int kk, int lane) {
    const int col   = colBase + (lane & 15);
    const int kbase = kk + ((lane < 16) ? 0 : 8);
    const _Float16* p = m + col * 64 + kbase;            // 16B aligned
    const v8h lo = *(const v8h*)(p);
    const v8h hi = *(const v8h*)(p + 16);
    return __builtin_shufflevector(lo, hi, 0,1,2,3,4,5,6,7,8,9,10,11,12,13,14,15);
}

// ---------------------------------------------------------------------------
// K1: per-(h,n) DSS coefficients.  w = exp(log_step)*Lambda.  argmax of the
// (linear-in-l) real exponent is L-1 if Re(w)>0 else 0.  s = sum_l exp(w(l-idx))
// via block reduction; a~ = (W/Lambda) * conj(s)/(|s|^2+eps).
// ---------------------------------------------------------------------------
__global__ void dss_coef_kernel(const float* __restrict__ W,
                                const float* __restrict__ log_step,
                                const float* __restrict__ lamRe,
                                const float* __restrict__ lamIm,
                                float* __restrict__ aRe, float* __restrict__ aIm,
                                float* __restrict__ wReA, float* __restrict__ wImA,
                                float* __restrict__ dIdxA)
{
    const int hn  = blockIdx.x;
    const int h   = hn >> 6;
    const int n   = hn & 63;
    const int tid = threadIdx.x;

    const float step = expf(log_step[h]);
    const float lr = lamRe[n], li = lamIm[n];
    const float wr = step * lr, wi = step * li;
    const float di = (wr > 0.f) ? (float)(LL - 1) : 0.f;

    float sre = 0.f, sim = 0.f;
    for (int l = tid; l < LL; l += 128) {
        const float d = (float)l - di;        // wr*d <= 0 always -> no overflow
        const float mag = expf(wr * d);
        float sn, cs; __sincosf(wi * d, &sn, &cs);
        sre += mag * cs;
        sim += mag * sn;
    }
    __shared__ float rr[128], ri[128];
    rr[tid] = sre; ri[tid] = sim;
    __syncthreads();
    for (int off = 64; off > 0; off >>= 1) {
        if (tid < off) { rr[tid] += rr[tid + off]; ri[tid] += ri[tid + off]; }
        __syncthreads();
    }
    if (tid == 0) {
        const float sr = rr[0], si = ri[0];
        const float den = sr * sr + si * si + 1e-7f;
        const float rcr =  sr / den;          // conj(s)/(|s|^2+eps)
        const float rci = -si / den;
        const float Wr = W[hn * 2 + 0], Wi = W[hn * 2 + 1];
        const float ldn = lr * lr + li * li;
        const float qr = (Wr * lr + Wi * li) / ldn;   // W / Lambda
        const float qi = (Wi * lr - Wr * li) / ldn;
        aRe[hn] = qr * rcr - qi * rci;
        aIm[hn] = qr * rci + qi * rcr;
        wReA[hn] = wr; wImA[hn] = wi; dIdxA[hn] = di;
    }
}

// ---------------------------------------------------------------------------
// K2: per-h tables.
//  T0[tau][rho] = (rho<=tau) ? K[tau-rho] : 0              (f16, WMMA B operand)
//  Exponent split with pc+qc+lc = -idx, each factor bounded:
//   Re(w)>0 : pc=-64, qc=0,  lc=-(L-C-1)=-4031
//   Re(w)<=0: pc=0,   qc=63, lc=-(C-1) =-63
//  P[h][n][tau] = a~ * exp(w*(tau+pc))   (f32)
//  Q[h][n][rho] = exp(w*(qc-rho))        (f16, |.|<=1, WMMA B operand, N-major)
//  Lam[h][d][n] = exp(w*(64d+lc)), d in [1,63], row 0 = 0   (f32)
// ---------------------------------------------------------------------------
__global__ void dss_table_kernel(const float* __restrict__ aRe, const float* __restrict__ aIm,
                                 const float* __restrict__ wReA, const float* __restrict__ wImA,
                                 const float* __restrict__ dIdxA,
                                 float* __restrict__ PRe, float* __restrict__ PIm,
                                 float* __restrict__ LRe, float* __restrict__ LIm,
                                 _Float16* __restrict__ Tm,
                                 _Float16* __restrict__ QRe, _Float16* __restrict__ QIm)
{
    const int h = blockIdx.x;
    const int tid = threadIdx.x;
    __shared__ float cAr[NN], cAi[NN], cWr[NN], cWi[NN], cDi[NN], Kd[CC];
    if (tid < NN) {
        const int hn = h * NN + tid;
        cAr[tid] = aRe[hn];  cAi[tid] = aIm[hn];
        cWr[tid] = wReA[hn]; cWi[tid] = wImA[hn];
        cDi[tid] = dIdxA[hn];
    }
    __syncthreads();
    if (tid < CC) {   // K[d] = Re sum_n a~ exp(w(d-idx))
        float acc = 0.f;
        for (int n = 0; n < NN; ++n) {
            const float d = (float)tid - cDi[n];
            const float mag = expf(cWr[n] * d);
            float sn, cs; __sincosf(cWi[n] * d, &sn, &cs);
            acc += cAr[n] * (mag * cs) - cAi[n] * (mag * sn);
        }
        Kd[tid] = acc;
    }
    __syncthreads();
    const size_t hb = (size_t)h * (CC * NN);
    for (int idx = tid; idx < CC * CC; idx += blockDim.x) {      // T0
        const int tau = idx >> 6, rho = idx & 63;
        Tm[hb + idx] = (_Float16)((rho <= tau) ? Kd[tau - rho] : 0.f);
    }
    for (int idx = tid; idx < NN * CC; idx += blockDim.x) {      // P[h][n][tau]
        const int n = idx >> 6, tau = idx & 63;
        const float pc  = (cWr[n] > 0.f) ? -64.f : 0.f;
        const float arg = (float)tau + pc;                        // Re(w)*arg <= 0
        const float mag = expf(cWr[n] * arg);
        float sn, cs; __sincosf(cWi[n] * arg, &sn, &cs);
        const float er = mag * cs, ei = mag * sn;
        PRe[hb + idx] = cAr[n] * er - cAi[n] * ei;
        PIm[hb + idx] = cAr[n] * ei + cAi[n] * er;
    }
    for (int idx = tid; idx < NC * NN; idx += blockDim.x) {      // Lam[h][delta][n]
        const int dl = idx >> 6, n = idx & 63;
        float outr = 0.f, outi = 0.f;
        if (dl > 0) {
            const float lc  = (cWr[n] > 0.f) ? -(float)(LL - CC - 1) : -(float)(CC - 1);
            const float arg = (float)(CC * dl) + lc;              // Re(w)*arg <= Re(w)
            const float ex  = fminf(cWr[n] * arg, 80.f);
            const float mag = expf(ex);
            float sn, cs; __sincosf(cWi[n] * arg, &sn, &cs);
            outr = mag * cs; outi = mag * sn;
        }
        LRe[hb + idx] = outr; LIm[hb + idx] = outi;
    }
    for (int idx = tid; idx < NN * CC; idx += blockDim.x) {      // Q[h][n][rho]  (N-major)
        const int n = idx >> 6, rho = idx & 63;
        const float qc  = (cWr[n] > 0.f) ? 0.f : (float)(CC - 1);
        const float arg = qc - (float)rho;                        // Re(w)*arg <= 0
        const float mag = expf(cWr[n] * arg);
        float sn, cs; __sincosf(cWi[n] * arg, &sn, &cs);
        QRe[hb + idx] = (_Float16)(mag * cs);
        QIm[hb + idx] = (_Float16)(mag * sn);
    }
}

// ---------------------------------------------------------------------------
// K3: fused per-(b,h) pipeline, 128 threads = 4 waves, 64KB dynamic LDS.
//  P1: V[j][n] = sum_rho U[j][rho] * Q[n][rho]          (2x WMMA f16, re/im)
//  P2: Z[i][n] = sum_{d>=1} Lam_d[n] * V[i-d][n]        (VALU, registers)
//  P2.5: Yfar[i][tau] = Re( sum_n Z[i][n] * P[n][tau] ) (VALU -> LDS)
//  P3: Y0 = U * T0^T (WMMA f16) ; out = Y0 + Yfar + D*u (single store)
// ---------------------------------------------------------------------------
__global__ void dss_main_kernel(const float* __restrict__ u, const float* __restrict__ D,
                                const float* __restrict__ PRe, const float* __restrict__ PIm,
                                const float* __restrict__ LRe, const float* __restrict__ LIm,
                                const _Float16* __restrict__ Tm,
                                const _Float16* __restrict__ QRe, const _Float16* __restrict__ QIm,
                                float* __restrict__ out)
{
    extern __shared__ char smem[];
    // phase-overlaid carve-up (64 KB total)
    float*     VtRe  = (float*)(smem);                 // P1 out / dead in P3
    float*     VtIm  = (float*)(smem + 16384);
    float*     Yfar  = (float*)(smem + 16384);         // P2.5 out (VtIm dead)
    _Float16*  Tsh   = (_Float16*)(smem);              // P3 (Vt dead)
    _Float16*  Ush2  = (_Float16*)(smem + 8192);       // P3
    _Float16*  Ush   = (_Float16*)(smem + 32768);      // P1
    _Float16*  QshRe = (_Float16*)(smem + 40960);      // P1
    _Float16*  QshIm = (_Float16*)(smem + 49152);      // P1
    float*     ZtRe  = (float*)(smem + 32768);         // P2 out (Ush/Qsh dead)
    float*     ZtIm  = (float*)(smem + 49152);

    const int tid  = threadIdx.x;
    const int bh   = blockIdx.x;
    const int h    = bh & (HH - 1);
    const size_t base = (size_t)bh * LL;
    const size_t hoff = (size_t)h * (CC * NN);
    const float  Dh   = D[h];

    const int lane    = tid & 31;
    const int wv      = tid >> 5;
    const int rowBase = wv * 16;
    const int r0      = (lane < 16) ? 0 : 8;
    const int col15   = lane & 15;

    if (tid == 0) {   // pull per-h tables toward the caches (global_prefetch_b8)
        __builtin_prefetch(PRe + hoff, 0, 3);
        __builtin_prefetch(PIm + hoff, 0, 3);
        __builtin_prefetch(LRe + hoff, 0, 3);
        __builtin_prefetch(LIm + hoff, 0, 3);
        __builtin_prefetch(Tm + hoff, 0, 3);
    }

    // ---- P1: stage U (f32->f16, b128 loads) + Q (b128 copies), WMMA V ----
    for (int idx = tid * 8; idx < NC * CC; idx += 128 * 8) {
        const v4f u0 = *(const v4f*)(u + base + idx);
        const v4f u1 = *(const v4f*)(u + base + idx + 4);
        v8h hv;
        hv[0] = (_Float16)u0[0]; hv[1] = (_Float16)u0[1];
        hv[2] = (_Float16)u0[2]; hv[3] = (_Float16)u0[3];
        hv[4] = (_Float16)u1[0]; hv[5] = (_Float16)u1[1];
        hv[6] = (_Float16)u1[2]; hv[7] = (_Float16)u1[3];
        *(v8h*)(Ush + idx) = hv;
    }
    for (int idx = tid * 8; idx < CC * NN; idx += 128 * 8) {
        *(v8h*)(QshRe + idx) = *(const v8h*)(QRe + hoff + idx);
        *(v8h*)(QshIm + idx) = *(const v8h*)(QIm + hoff + idx);
    }
    __syncthreads();
    {
        const v16h a0 = load_a_frag(Ush, rowBase, 0, lane);
        const v16h a1 = load_a_frag(Ush, rowBase, 32, lane);
        for (int tg = 0; tg < 4; ++tg) {
            v8f accR = {0.f,0.f,0.f,0.f,0.f,0.f,0.f,0.f};
            v8f accI = {0.f,0.f,0.f,0.f,0.f,0.f,0.f,0.f};
            const v16h br0 = load_b_frag(QshRe, tg * 16, 0, lane);
            const v16h br1 = load_b_frag(QshRe, tg * 16, 32, lane);
            const v16h bi0 = load_b_frag(QshIm, tg * 16, 0, lane);
            const v16h bi1 = load_b_frag(QshIm, tg * 16, 32, lane);
            accR = __builtin_amdgcn_wmma_f32_16x16x32_f16(false, a0, false, br0, (short)0, accR, false, false);
            accR = __builtin_amdgcn_wmma_f32_16x16x32_f16(false, a1, false, br1, (short)0, accR, false, false);
            accI = __builtin_amdgcn_wmma_f32_16x16x32_f16(false, a0, false, bi0, (short)0, accI, false, false);
            accI = __builtin_amdgcn_wmma_f32_16x16x32_f16(false, a1, false, bi1, (short)0, accI, false, false);
#pragma unroll
            for (int r = 0; r < 8; ++r) {
                const int m = rowBase + r0 + r;
                VtRe[m * 64 + tg * 16 + col15] = accR[r];
                VtIm[m * 64 + tg * 16 + col15] = accI[r];
            }
        }
    }
    __syncthreads();

    // ---- P2: Z[i][n] = sum_{d=1..63, i-d>=0} Lam_d[n] * V[i-d][n] ----
    {
        const int n    = tid & 63;
        const int half = tid >> 6;
        const int i0   = half * 32;
        float zr[32], zi[32];
#pragma unroll
        for (int ii = 0; ii < 32; ++ii) { zr[ii] = 0.f; zi[ii] = 0.f; }
        for (int dl = 1; dl < NC; ++dl) {
            const float lr = LRe[hoff + dl * 64 + n];
            const float li = LIm[hoff + dl * 64 + n];
#pragma unroll 4
            for (int ii = 0; ii < 32; ++ii) {
                const int j = i0 + ii - dl;
                if (j >= 0) {
                    const float vr = VtRe[j * 64 + n];
                    const float vi = VtIm[j * 64 + n];
                    zr[ii] += lr * vr - li * vi;
                    zi[ii] += lr * vi + li * vr;
                }
            }
        }
#pragma unroll
        for (int ii = 0; ii < 32; ++ii) {
            ZtRe[(i0 + ii) * 64 + n] = zr[ii];
            ZtIm[(i0 + ii) * 64 + n] = zi[ii];
        }
    }
    __syncthreads();

    // ---- P2.5: Yfar[i][tau] = Re( sum_n Z[i][n] P[n][tau] ) ----
    {
        const int tau  = tid & 63;
        const int half = tid >> 6;
        const int i0   = half * 32;
        float acc[32];
#pragma unroll
        for (int ii = 0; ii < 32; ++ii) acc[ii] = 0.f;
        for (int n = 0; n < NN; ++n) {
            const float pr = PRe[hoff + n * 64 + tau];
            const float pi = PIm[hoff + n * 64 + tau];
#pragma unroll 4
            for (int ii = 0; ii < 32; ++ii) {
                acc[ii] += ZtRe[(i0 + ii) * 64 + n] * pr - ZtIm[(i0 + ii) * 64 + n] * pi;
            }
        }
#pragma unroll
        for (int ii = 0; ii < 32; ++ii) Yfar[(i0 + ii) * 64 + tau] = acc[ii];
    }
    __syncthreads();

    // ---- P3: Y0 = U * T0^T via WMMA; fused epilogue out = Y0 + Yfar + D*u ----
    for (int idx = tid * 8; idx < CC * CC; idx += 128 * 8) {
        *(v8h*)(Tsh + idx) = *(const v8h*)(Tm + hoff + idx);
    }
    for (int idx = tid * 8; idx < NC * CC; idx += 128 * 8) {
        const v4f u0 = *(const v4f*)(u + base + idx);
        const v4f u1 = *(const v4f*)(u + base + idx + 4);
        v8h hv;
        hv[0] = (_Float16)u0[0]; hv[1] = (_Float16)u0[1];
        hv[2] = (_Float16)u0[2]; hv[3] = (_Float16)u0[3];
        hv[4] = (_Float16)u1[0]; hv[5] = (_Float16)u1[1];
        hv[6] = (_Float16)u1[2]; hv[7] = (_Float16)u1[3];
        *(v8h*)(Ush2 + idx) = hv;
    }
    __syncthreads();
    {
        const v16h a0 = load_a_frag(Ush2, rowBase, 0, lane);
        const v16h a1 = load_a_frag(Ush2, rowBase, 32, lane);
        for (int tg = 0; tg < 4; ++tg) {
            v8f acc = {0.f,0.f,0.f,0.f,0.f,0.f,0.f,0.f};
            const v16h b0 = load_b_frag(Tsh, tg * 16, 0, lane);
            const v16h b1 = load_b_frag(Tsh, tg * 16, 32, lane);
            acc = __builtin_amdgcn_wmma_f32_16x16x32_f16(false, a0, false, b0, (short)0, acc, false, false);
            acc = __builtin_amdgcn_wmma_f32_16x16x32_f16(false, a1, false, b1, (short)0, acc, false, false);
#pragma unroll
            for (int r = 0; r < 8; ++r) {
                const int i = rowBase + r0 + r;
                const int l = i * 64 + tg * 16 + col15;
                out[base + l] = acc[r] + Yfar[l] + Dh * u[base + l];
            }
        }
    }
}

// ---------------------------------------------------------------------------
extern "C" void kernel_launch(void* const* d_in, const int* in_sizes, int n_in,
                              void* d_out, int out_size, void* d_ws, size_t ws_size,
                              hipStream_t stream)
{
    (void)in_sizes; (void)n_in; (void)out_size; (void)ws_size;
    const float* u   = (const float*)d_in[0];
    const float* W   = (const float*)d_in[1];
    const float* D   = (const float*)d_in[2];
    const float* ls  = (const float*)d_in[3];
    const float* lre = (const float*)d_in[4];
    const float* lim = (const float*)d_in[5];
    float* out = (float*)d_out;

    const size_t HN  = (size_t)HH * NN;        // 32768
    const size_t TBL = (size_t)HH * CC * NN;   // 2M entries per table
    float* w   = (float*)d_ws;
    float* aRe = w;             float* aIm = aRe + HN;
    float* wRe = aIm + HN;      float* wIm = wRe + HN;
    float* dIx = wIm + HN;
    float* PRe = dIx + HN;      float* PIm = PRe + TBL;
    float* LRe = PIm + TBL;     float* LIm = LRe + TBL;
    _Float16* Tm  = (_Float16*)(LIm + TBL);
    _Float16* QRe = Tm + TBL;
    _Float16* QIm = QRe + TBL;

    dss_coef_kernel<<<dim3(HH * NN), dim3(128), 0, stream>>>(W, ls, lre, lim,
                                                             aRe, aIm, wRe, wIm, dIx);
    dss_table_kernel<<<dim3(HH), dim3(256), 0, stream>>>(aRe, aIm, wRe, wIm, dIx,
                                                         PRe, PIm, LRe, LIm, Tm, QRe, QIm);
    dss_main_kernel<<<dim3(BB * HH), dim3(128), 65536, stream>>>(u, D, PRe, PIm, LRe, LIm,
                                                                 Tm, QRe, QIm, out);
}
